// DIRLNet_1443109011972
// MI455X (gfx1250) — matrix operand order repaired and checked by reading, not compiled, not run on hardware
//
#include <hip/hip_runtime.h>
#include <hip/hip_bf16.h>

// ---------------------------------------------------------------------------
// MI455X (gfx1250) implementation of the NNConv+GRU DIRL net.
//
// Roofline: materializing per-edge 32x32 weights = 1.02GB -> ~4GB HBM traffic
// (~175us @ 23.3TB/s).  Instead we recompute ew = he@We2 every iteration with
// v_wmma_f32_16x16x32_bf16 and keep he (32MB f32) + We2^T (64KB bf16) L2-
// resident.  Key trick: the per-h scalar out[src][e,h] is folded into the A
// operand (one scalar per lane since A rows map 1:1 to lanes), so every WMMA
// accumulates straight into the message accumulator (D->C chaining, no hazard
// NOPs) and no VALU ever consumes a WMMA result inside the hot loop.  The
// bias term initializes C via one extra WMMA pair per tile (out_src @ be2^T).
// Each wave processes 4 tiles so each B fragment feeds 4 WMMAs.  Scatter uses
// f32 global atomics (same semantics as jax segment_sum).  Node-level GEMMs
// (~0.7 GFLOP/iter) stay f32 VALU for fidelity.
// ---------------------------------------------------------------------------

#define N_NODES 50000
#define N_EDGES 250000
#define N_GRAPH 512
#define HDIM    32

typedef __attribute__((ext_vector_type(16))) __bf16 bf16x16;
typedef __attribute__((ext_vector_type(8)))  float  f32x8;

__device__ __forceinline__ float sigmoidf_(float x) {
    return 1.0f / (1.0f + __expf(-x));
}

// ------------------------- generic zero fill -------------------------------
__global__ void k_zero(float* p, int n) {
    int i = blockIdx.x * blockDim.x + threadIdx.x;
    if (i < n) p[i] = 0.0f;
}

// ---------------- We2^T + be2^T -> bf16, column-major ----------------------
// We2 is [32, 1024] row-major (k-major).  we2t[col][k] bf16, 64B per column,
// so a WMMA B-fragment (lane half h: K 8h..8h+7 and 16+8h..16+8h+7 of one column) is two b128 loads.
// be2 is [32h, 32o]; be2t[o][h] bf16 with the same fragment layout.
__global__ void k_prep_w(const float* We2, const float* be2,
                         __bf16* we2t, __bf16* be2t) {
    int c = blockIdx.x * blockDim.x + threadIdx.x;
    if (c >= 1024) return;
    __bf16 tmp[32];
#pragma unroll
    for (int k = 0; k < 32; ++k)
        tmp[k] = (__bf16)We2[k * 1024 + c];
    uint4* dst = (uint4*)(we2t + (size_t)c * 32);
    const uint4* src = (const uint4*)tmp;
#pragma unroll
    for (int j = 0; j < 4; ++j) dst[j] = src[j];
    // be2t[o*32+h] = be2[h*32+o]
    be2t[c] = (__bf16)be2[(c & 31) * 32 + (c >> 5)];
}

// ------------------------- edge MLP layer 1 -> he (f32) --------------------
// he[e][k] = relu(edge_attr[e,:16] @ We1[16,32] + be1)[k]
__global__ __launch_bounds__(256) void k_edge_hidden(
    const float* __restrict__ ea, const float* __restrict__ We1,
    const float* __restrict__ be1, float* __restrict__ he) {
    __shared__ float sW[16 * 32];
    __shared__ float sb[32];
    for (int i = threadIdx.x; i < 512; i += blockDim.x) sW[i] = We1[i];
    if (threadIdx.x < 32) sb[threadIdx.x] = be1[threadIdx.x];
    __syncthreads();
    int e = blockIdx.x * blockDim.x + threadIdx.x;
    if (e >= N_EDGES) return;
    float a[16];
    const float4* ap = (const float4*)(ea + (size_t)e * 16);
#pragma unroll
    for (int j = 0; j < 4; ++j) {
        float4 v = ap[j];
        a[4 * j + 0] = v.x; a[4 * j + 1] = v.y; a[4 * j + 2] = v.z; a[4 * j + 3] = v.w;
    }
    float* dst = he + (size_t)e * 32;
#pragma unroll
    for (int o = 0; o < 32; ++o) {
        float s = sb[o];
#pragma unroll
        for (int k = 0; k < 16; ++k) s += a[k] * sW[k * 32 + o];
        dst[o] = fmaxf(s, 0.0f);
    }
}

// ------------------------- lin0: h = relu(x @ W0 + b0) ---------------------
__global__ __launch_bounds__(256) void k_lin0(
    const float* __restrict__ x, const float* __restrict__ W0,
    const float* __restrict__ b0, float* __restrict__ h) {
    __shared__ float sW[32 * 32];
    __shared__ float sb[32];
    for (int i = threadIdx.x; i < 1024; i += blockDim.x) sW[i] = W0[i];
    if (threadIdx.x < 32) sb[threadIdx.x] = b0[threadIdx.x];
    __syncthreads();
    int n = blockIdx.x * blockDim.x + threadIdx.x;
    if (n >= N_NODES) return;
    float xr[32];
    const float4* xp = (const float4*)(x + (size_t)n * 32);
#pragma unroll
    for (int j = 0; j < 8; ++j) {
        float4 v = xp[j];
        xr[4 * j + 0] = v.x; xr[4 * j + 1] = v.y; xr[4 * j + 2] = v.z; xr[4 * j + 3] = v.w;
    }
    float* hr = h + (size_t)n * 32;
#pragma unroll
    for (int o = 0; o < 32; ++o) {
        float s = sb[o];
#pragma unroll
        for (int k = 0; k < 32; ++k) s += xr[k] * sW[k * 32 + o];
        hr[o] = fmaxf(s, 0.0f);
    }
}

// ------------------------- in-degree counts --------------------------------
__global__ void k_count(const int* __restrict__ ei, float* __restrict__ cnt) {
    int e = blockIdx.x * blockDim.x + threadIdx.x;
    if (e < N_EDGES) atomicAdd(&cnt[ei[N_EDGES + e]], 1.0f);
}

// ------------------------- fused WMMA edge-message kernel ------------------
#define TILES_PER_WAVE 4
#define OS_STRIDE 17            // padded transposed-out_src row stride (f32)
#define TILE_BUF  544           // max(32*OS_STRIDE, 512)

__global__ __launch_bounds__(128) void k_edge_msg(
    const float* __restrict__ he, const __bf16* __restrict__ we2t,
    const __bf16* __restrict__ be2t, const int* __restrict__ ei,
    const float* __restrict__ h, float* __restrict__ agg,
    int numTiles, int numGroups) {
    __shared__ float s_tile[4][TILES_PER_WAVE][TILE_BUF]; // os_t, then msg staging

    const int lane  = threadIdx.x & 31;
    const int wave  = threadIdx.x >> 5;
    const int half  = lane >> 4;           // 0: lanes 0-15, 1: lanes 16-31
    const int l16   = lane & 15;
    const int gwave = blockIdx.x * (blockDim.x >> 5) + wave;
    const int nwav  = gridDim.x * (blockDim.x >> 5);
    const int mbase = half * 8;
    float (*myt)[TILE_BUF] = s_tile[wave];

    union BF { bf16x16 v; uint4 q[2]; };

    // constant be2^T B fragments (cols o 0-15 and 16-31), held in registers
    BF bb0, bb1;
    {
        const uint4* p0 = (const uint4*)(be2t + ((size_t)l16 * 32));
        bb0.q[0] = p0[half]; bb0.q[1] = p0[2 + half];
        const uint4* p1 = (const uint4*)(be2t + ((size_t)(l16 + 16) * 32));
        bb1.q[0] = p1[half]; bb1.q[1] = p1[2 + half];
    }

    for (int grp = gwave; grp < numGroups; grp += nwav) {
        const int tbase = grp * TILES_PER_WAVE;

        int   eclamp[TILES_PER_WAVE];
        bool  valid[TILES_PER_WAVE];
        float hev[TILES_PER_WAVE][16];        // he row (f32), A-swizzle K-order
        f32x8 msg0[TILES_PER_WAVE], msg1[TILES_PER_WAVE];

#pragma unroll
        for (int t = 0; t < TILES_PER_WAVE; ++t) {
            const int tile = tbase + t;
            valid[t] = (tile < numTiles);
            const int e = valid[t] ? tile * 16 + l16 : l16;   // lane's row (edge)
            eclamp[t] = e;

            // out[src] row: A-swizzle subsets.  half0 -> K{0-7,16-23} (float4
            // idx 0,1,4,5); half1 -> K{8-15,24-31} (idx 2,3,6,7).
            const int srcn = ei[e];
            const float4* hp = (const float4*)(h + (size_t)srcn * 32);
            float4 o0 = hp[half * 2 + 0], o1 = hp[half * 2 + 1];
            float4 o2 = hp[half * 2 + 4], o3 = hp[half * 2 + 5];
            float osv[16] = { o0.x,o0.y,o0.z,o0.w, o1.x,o1.y,o1.z,o1.w,
                              o2.x,o2.y,o2.z,o2.w, o3.x,o3.y,o3.z,o3.w };
            // transposed out_src into LDS: os_t[k][e], padded stride
#pragma unroll
            for (int j = 0; j < 16; ++j) {
                const int K = (j < 8 ? j : j + 8) + half * 8;
                myt[t][K * OS_STRIDE + l16] = osv[j];
            }
            // bias init: C = out_src_tile @ be2^T  (2 WMMAs, C=0)
            union { bf16x16 v; __bf16 b[16]; } aos;
#pragma unroll
            for (int j = 0; j < 16; ++j) aos.b[j] = (__bf16)osv[j];
            f32x8 z = {};
            msg0[t] = __builtin_amdgcn_wmma_f32_16x16x32_bf16(
                false, aos.v, false, bb0.v, (short)0, z, false, false);
            msg1[t] = __builtin_amdgcn_wmma_f32_16x16x32_bf16(
                false, aos.v, false, bb1.v, (short)0, z, false, false);

            // he row (f32), same A-swizzle subsets
            const float4* gp = (const float4*)(he + (size_t)e * 32);
            float4 g0 = gp[half * 2 + 0], g1 = gp[half * 2 + 1];
            float4 g2 = gp[half * 2 + 4], g3 = gp[half * 2 + 5];
            hev[t][0] = g0.x; hev[t][1]  = g0.y; hev[t][2]  = g0.z; hev[t][3]  = g0.w;
            hev[t][4] = g1.x; hev[t][5]  = g1.y; hev[t][6]  = g1.z; hev[t][7]  = g1.w;
            hev[t][8] = g2.x; hev[t][9]  = g2.y; hev[t][10] = g2.z; hev[t][11] = g2.w;
            hev[t][12] = g3.x; hev[t][13] = g3.y; hev[t][14] = g3.z; hev[t][15] = g3.w;
        }

        for (int hh = 0; hh < 32; ++hh) {
            // B fragments: We2^T columns [hh*32, hh*32+32), shared by 4 tiles
            BF b0f, b1f;
            {
                const int col0 = hh * 32 + l16;
                const uint4* bp0 = (const uint4*)(we2t + (size_t)col0 * 32);
                b0f.q[0] = bp0[half]; b0f.q[1] = bp0[2 + half];
                const uint4* bp1 = (const uint4*)(we2t + (size_t)(col0 + 16) * 32);
                b1f.q[0] = bp1[half]; b1f.q[1] = bp1[2 + half];
            }
#pragma unroll
            for (int t = 0; t < TILES_PER_WAVE; ++t) {
                // fold per-h scalar into A: one scalar per lane (row M = l16)
                const float s = myt[t][hh * OS_STRIDE + l16];
                union { bf16x16 v; __bf16 b[16]; } asc;
#pragma unroll
                for (int j = 0; j < 16; ++j) asc.b[j] = (__bf16)(s * hev[t][j]);
                msg0[t] = __builtin_amdgcn_wmma_f32_16x16x32_bf16(
                    false, asc.v, false, b0f.v, (short)0, msg0[t], false, false);
                msg1[t] = __builtin_amdgcn_wmma_f32_16x16x32_bf16(
                    false, asc.v, false, b1f.v, (short)0, msg1[t], false, false);
            }
        }

        // ---- stage msg tiles to LDS (reuse buffers), scatter with atomics ---
#pragma unroll
        for (int t = 0; t < TILES_PER_WAVE; ++t) {
#pragma unroll
            for (int r = 0; r < 8; ++r) {
                const int M = mbase + r;
                myt[t][M * 32 + l16]      = msg0[t][r];
                myt[t][M * 32 + 16 + l16] = msg1[t][r];
            }
            if (valid[t]) {
                const int dstn = ei[N_EDGES + eclamp[t]];
                float* aggrow = agg + (size_t)dstn * 32 + half * 16;
                const float* mr = &myt[t][l16 * 32 + half * 16];
#pragma unroll
                for (int j = 0; j < 16; ++j) atomicAdd(aggrow + j, mr[j]);
            }
        }
    }
}

// ------------------------- fused NNConv-root + GRU node update -------------
__global__ __launch_bounds__(256) void k_node_update(
    const float* __restrict__ agg, const float* __restrict__ cnt,
    float* __restrict__ h,
    const float* __restrict__ Wroot, const float* __restrict__ bconv,
    const float* __restrict__ Wih, const float* __restrict__ bih,
    const float* __restrict__ Whh, const float* __restrict__ bhh) {
    __shared__ float sWr[1024], sWi[3072], sWh[3072];
    __shared__ float sbc[32], sbi[96], sbh[96];
    for (int i = threadIdx.x; i < 1024; i += blockDim.x) sWr[i] = Wroot[i];
    for (int i = threadIdx.x; i < 3072; i += blockDim.x) { sWi[i] = Wih[i]; sWh[i] = Whh[i]; }
    if (threadIdx.x < 96) { sbi[threadIdx.x] = bih[threadIdx.x]; sbh[threadIdx.x] = bhh[threadIdx.x]; }
    if (threadIdx.x < 32) sbc[threadIdx.x] = bconv[threadIdx.x];
    __syncthreads();

    int n = blockIdx.x * blockDim.x + threadIdx.x;
    if (n >= N_NODES) return;

    float hr[32], m[32];
    float* hrow = h + (size_t)n * 32;
#pragma unroll
    for (int j = 0; j < 8; ++j) {
        float4 v = ((const float4*)hrow)[j];
        hr[4 * j + 0] = v.x; hr[4 * j + 1] = v.y; hr[4 * j + 2] = v.z; hr[4 * j + 3] = v.w;
    }
    const float inv = 1.0f / fmaxf(cnt[n], 1.0f);
    const float* arow = agg + (size_t)n * 32;
#pragma unroll
    for (int o = 0; o < 32; ++o) {
        float s = arow[o] * inv + sbc[o];
#pragma unroll
        for (int k = 0; k < 32; ++k) s += hr[k] * sWr[k * 32 + o];
        m[o] = fmaxf(s, 0.0f);
    }
    float hn[32];
#pragma unroll
    for (int j = 0; j < 32; ++j) {
        float gr = sbi[j], pr = sbh[j];
        float gz = sbi[32 + j], pz = sbh[32 + j];
        float gn = sbi[64 + j], pn = sbh[64 + j];
#pragma unroll
        for (int k = 0; k < 32; ++k) {
            gr += m[k] * sWi[j * 32 + k];        pr += hr[k] * sWh[j * 32 + k];
            gz += m[k] * sWi[(32 + j) * 32 + k]; pz += hr[k] * sWh[(32 + j) * 32 + k];
            gn += m[k] * sWi[(64 + j) * 32 + k]; pn += hr[k] * sWh[(64 + j) * 32 + k];
        }
        const float r = sigmoidf_(gr + pr);
        const float z = sigmoidf_(gz + pz);
        const float nn = tanhf(gn + r * pn);
        hn[j] = (1.0f - z) * nn + z * hr[j];
    }
#pragma unroll
    for (int j = 0; j < 8; ++j) {
        float4 v;
        v.x = hn[4 * j + 0]; v.y = hn[4 * j + 1]; v.z = hn[4 * j + 2]; v.w = hn[4 * j + 3];
        ((float4*)hrow)[j] = v;
    }
}

// ------------------------- graph pooling -----------------------------------
__global__ void k_pool(const float* __restrict__ h, const int* __restrict__ batch,
                       float* __restrict__ pooled) {
    int idx = blockIdx.x * blockDim.x + threadIdx.x;
    if (idx >= N_NODES * 32) return;
    int n = idx >> 5, o = idx & 31;
    atomicAdd(&pooled[batch[n] * 32 + o], h[idx]);
}
__global__ void k_gcnt(const int* __restrict__ batch, float* __restrict__ gcnt) {
    int n = blockIdx.x * blockDim.x + threadIdx.x;
    if (n < N_NODES) atomicAdd(&gcnt[batch[n]], 1.0f);
}

// ------------------------- output heads ------------------------------------
__global__ __launch_bounds__(512) void k_heads(
    const float* __restrict__ pooled, const float* __restrict__ gcnt,
    const float* __restrict__ W1a, const float* __restrict__ b1a,
    const float* __restrict__ W1b, const float* __restrict__ b1b,
    const float* __restrict__ W2a, const float* __restrict__ b2a,
    const float* __restrict__ W2b, const float* __restrict__ b2b,
    float* __restrict__ out) {
    __shared__ float sW1a[1024], sW2a[1024], sW1b[32], sW2b[64];
    __shared__ float sb1a[32], sb2a[32], sb1b, sb2b0, sb2b1;
    for (int i = threadIdx.x; i < 1024; i += blockDim.x) { sW1a[i] = W1a[i]; sW2a[i] = W2a[i]; }
    if (threadIdx.x < 64) sW2b[threadIdx.x] = W2b[threadIdx.x];
    if (threadIdx.x < 32) {
        sW1b[threadIdx.x] = W1b[threadIdx.x];
        sb1a[threadIdx.x] = b1a[threadIdx.x];
        sb2a[threadIdx.x] = b2a[threadIdx.x];
    }
    if (threadIdx.x == 0) { sb1b = b1b[0]; sb2b0 = b2b[0]; sb2b1 = b2b[1]; }
    __syncthreads();

    int g = threadIdx.x;
    if (g >= N_GRAPH) return;
    const float inv = 1.0f / fmaxf(gcnt[g], 1.0f);
    float p[32];
#pragma unroll
    for (int k = 0; k < 32; ++k) p[k] = pooled[g * 32 + k] * inv;

    float lab = sb1b;
    {
        float t[32];
#pragma unroll
        for (int o = 0; o < 32; ++o) {
            float s = sb1a[o];
#pragma unroll
            for (int k = 0; k < 32; ++k) s += p[k] * sW1a[k * 32 + o];
            t[o] = fmaxf(s, 0.0f);
        }
#pragma unroll
        for (int k = 0; k < 32; ++k) lab += t[k] * sW1b[k];
    }
    out[g] = lab;

    float d0 = sb2b0, d1 = sb2b1;
    {
        float t[32];
#pragma unroll
        for (int o = 0; o < 32; ++o) {
            float s = sb2a[o];
#pragma unroll
            for (int k = 0; k < 32; ++k) s += p[k] * sW2a[k * 32 + o];
            t[o] = fmaxf(s, 0.0f);
        }
#pragma unroll
        for (int k = 0; k < 32; ++k) { d0 += t[k] * sW2b[k * 2]; d1 += t[k] * sW2b[k * 2 + 1]; }
    }
    const float mx  = fmaxf(d0, d1);
    const float lse = mx + logf(__expf(d0 - mx) + __expf(d1 - mx));
    out[N_GRAPH + g * 2 + 0] = d0 - lse;
    out[N_GRAPH + g * 2 + 1] = d1 - lse;
}

// ---------------------------------------------------------------------------
extern "C" void kernel_launch(void* const* d_in, const int* in_sizes, int n_in,
                              void* d_out, int out_size, void* d_ws, size_t ws_size,
                              hipStream_t stream) {
    const float* x         = (const float*)d_in[0];
    const float* edge_attr = (const float*)d_in[1];
    const int*   edge_idx  = (const int*)  d_in[2];
    const int*   batch     = (const int*)  d_in[3];
    // d_in[4] = alpha (identity in forward)
    const float* W0    = (const float*)d_in[5];
    const float* b0    = (const float*)d_in[6];
    const float* We1   = (const float*)d_in[7];
    const float* be1   = (const float*)d_in[8];
    const float* We2   = (const float*)d_in[9];
    const float* be2   = (const float*)d_in[10];
    const float* Wroot = (const float*)d_in[11];
    const float* bconv = (const float*)d_in[12];
    const float* Wih   = (const float*)d_in[13];
    const float* bih   = (const float*)d_in[14];
    const float* Whh   = (const float*)d_in[15];
    const float* bhh   = (const float*)d_in[16];
    const float* W1a   = (const float*)d_in[17];
    const float* b1a   = (const float*)d_in[18];
    const float* W1b   = (const float*)d_in[19];
    const float* b1b   = (const float*)d_in[20];
    const float* W2a   = (const float*)d_in[21];
    const float* b2a   = (const float*)d_in[22];
    const float* W2b   = (const float*)d_in[23];
    const float* b2b   = (const float*)d_in[24];
    float* out = (float*)d_out;

    char* ws = (char*)d_ws;
    float*  he     = (float*) (ws + 0);                 // E*32 f32  = 32,000,000 B
    __bf16* we2t   = (__bf16*)(ws + 32000000);          // 1024*32 bf16 = 65,536 B
    __bf16* be2t   = (__bf16*)(ws + 32065536);          // 1024 bf16    =  2,048 B
    float*  h      = (float*) (ws + 32067584);          // N*32 f32 = 6,400,000 B
    float*  agg    = (float*) (ws + 38467584);          // N*32 f32
    float*  cnt    = (float*) (ws + 44867584);          // N f32
    float*  pooled = (float*) (ws + 45067776);          // G*32 f32
    float*  gcnt   = (float*) (ws + 45133312);          // G f32
    (void)ws_size; (void)in_sizes; (void)n_in; (void)out_size;

    const int numTiles  = N_EDGES / 16;                       // 15625
    const int numGroups = (numTiles + TILES_PER_WAVE - 1) / TILES_PER_WAVE; // 3907

    // --- one-time prep ----------------------------------------------------
    k_prep_w<<<4, 256, 0, stream>>>(We2, be2, we2t, be2t);
    k_edge_hidden<<<(N_EDGES + 255) / 256, 256, 0, stream>>>(edge_attr, We1, be1, he);
    k_lin0<<<(N_NODES + 255) / 256, 256, 0, stream>>>(x, W0, b0, h);
    k_zero<<<(N_NODES + 255) / 256, 256, 0, stream>>>(cnt, N_NODES);
    k_count<<<(N_EDGES + 255) / 256, 256, 0, stream>>>(edge_idx, cnt);

    // --- 3 message-passing iterations -------------------------------------
    const int msgBlocks = (numGroups + 3) / 4;                // 4 waves / block
    for (int it = 0; it < 3; ++it) {
        k_zero<<<(N_NODES * 32 + 255) / 256, 256, 0, stream>>>(agg, N_NODES * 32);
        k_edge_msg<<<msgBlocks, 128, 0, stream>>>(he, we2t, be2t, edge_idx, h, agg,
                                                  numTiles, numGroups);
        k_node_update<<<(N_NODES + 255) / 256, 256, 0, stream>>>(
            agg, cnt, h, Wroot, bconv, Wih, bih, Whh, bhh);
    }

    // --- pooling + heads ---------------------------------------------------
    k_zero<<<(N_GRAPH * 32 + 255) / 256, 256, 0, stream>>>(pooled, N_GRAPH * 32);
    k_zero<<<(N_GRAPH + 255) / 256, 256, 0, stream>>>(gcnt, N_GRAPH);
    k_pool<<<(N_NODES * 32 + 255) / 256, 256, 0, stream>>>(h, batch, pooled);
    k_gcnt<<<(N_NODES + 255) / 256, 256, 0, stream>>>(batch, gcnt);
    k_heads<<<1, 512, 0, stream>>>(pooled, gcnt, W1a, b1a, W1b, b1b,
                                   W2a, b2a, W2b, b2b, out);
}
